// MultiHeadAttention_32323923870543
// MI455X (gfx1250) — matrix operand order repaired and checked by reading, not compiled, not run on hardware
//
#include <hip/hip_runtime.h>
#include <hip/hip_bf16.h>

typedef __attribute__((ext_vector_type(16))) __bf16 v16bf;
typedef __attribute__((ext_vector_type(8)))  __bf16 v8bf;
typedef __attribute__((ext_vector_type(8)))  float  v8f;
typedef __attribute__((ext_vector_type(4)))  float  v4f;

#define BATCH   2
#define S_LEN   2048
#define D_MODEL 1024
#define N_HEADS 16
#define D_HEAD  64
// reference: scale = sqrt(D_MODEL) = 32 -> multiply scores by 1/32
#define INV_SCALE 0.03125f
// LDS row pitch (floats): 2048 + 4 rotates banks by 4 per row -> conflict-free
#define PITCH   2052

__device__ __forceinline__ __bf16 f2bf(float x) { return (__bf16)x; }

// CDNA5 16-bit A fragment (16x32), order used here: lane half selects K base (0 / 16),
// element e of the v16bf covers K = akb + e  -> one contiguous 16-element run.
// 16-bit B fragment (32x16), same order used here: lane half selects K base (0 / 16),
// element e covers K = bkb + e  -> one contiguous 16-element run.

// ---------------------------------------------------------------------------
// Kernel 1: fused Q/K/V projection  Y = X @ W^T  (fp32 in, bf16 out).
// Q,K stored row-major [4096,1024]; V stored per-head TRANSPOSED:
//   Vt[((b*16+h)*64 + dh) * 2048 + s2]  (so P@V B-fragments are contiguous).
// One wave = one 16x16 output tile, K loop in steps of 32.
// ---------------------------------------------------------------------------
__global__ __launch_bounds__(256)
void qkv_proj_kernel(const float* __restrict__ X,
                     const float* __restrict__ Wq,
                     const float* __restrict__ Wk,
                     const float* __restrict__ Wv,
                     __bf16* __restrict__ Q,
                     __bf16* __restrict__ K,
                     __bf16* __restrict__ Vt) {
  const int lane = threadIdx.x & 31;
  const int wv   = threadIdx.x >> 5;

  const int M  = BATCH * S_LEN;                   // 4096
  const int Kd = D_MODEL;
  const int N  = D_MODEL;
  const int tilesN      = N / 16;                 // 64
  const int tilesPerMat = (M / 16) * tilesN;      // 16384

  int gw    = blockIdx.x * 8 + wv;
  int which = gw / tilesPerMat;
  int rem   = gw - which * tilesPerMat;
  int tm    = rem / tilesN;
  int tn    = rem - tm * tilesN;

  const float* W = (which == 0) ? Wq : (which == 1) ? Wk : Wv;

  const int m   = tm * 16 + (lane & 15);
  const int n   = tn * 16 + (lane & 15);
  const int akb = (lane < 16) ? 0 : 16;   // A fragment K base
  const int bkb = (lane < 16) ? 0 : 16;   // B fragment K base

  const float* xrow = X + (size_t)m * Kd;
  const float* wrow = W + (size_t)n * Kd;

  v8f acc = {};
  for (int ks = 0; ks < Kd / 32; ++ks) {
    const float* ap = xrow + ks * 32 + akb;
    v4f x0 = *(const v4f*)(ap);
    v4f x1 = *(const v4f*)(ap + 4);
    v4f x2 = *(const v4f*)(ap + 8);
    v4f x3 = *(const v4f*)(ap + 12);
    const float* bp = wrow + ks * 32 + bkb;
    v4f w0 = *(const v4f*)(bp);
    v4f w1 = *(const v4f*)(bp + 4);
    v4f w2 = *(const v4f*)(bp + 8);
    v4f w3 = *(const v4f*)(bp + 12);
    v16bf a, bfrag;
#pragma unroll
    for (int i = 0; i < 4; ++i) {
      a[i]          = f2bf(x0[i]);
      a[4 + i]      = f2bf(x1[i]);
      a[8 + i]      = f2bf(x2[i]);
      a[12 + i]     = f2bf(x3[i]);
      bfrag[i]      = f2bf(w0[i]);
      bfrag[4 + i]  = f2bf(w1[i]);
      bfrag[8 + i]  = f2bf(w2[i]);
      bfrag[12 + i] = f2bf(w3[i]);
    }
    acc = __builtin_amdgcn_wmma_f32_16x16x32_bf16(false, a, false, bfrag,
                                                  (short)0, acc, false, false);
  }

  if (which == 2) {
    // Per-head transposed V store: Vh[s2][dh] -> Vt[dh][s2]
    const int nh = n >> 6;   // which 64-wide block of the model dim
    const int dh = n & 63;
#pragma unroll
    for (int v = 0; v < 8; ++v) {
      int row = tm * 16 + v + ((lane < 16) ? 0 : 8);   // global M row
      int b_  = row >> 11;
      int s   = row & 2047;
      int h_  = s >> 7;
      int s2  = ((s & 127) << 4) | nh;
      Vt[(((size_t)((b_ * N_HEADS + h_) * 64 + dh)) << 11) + s2] = f2bf(acc[v]);
    }
  } else {
    __bf16* Y = (which == 0) ? Q : K;
#pragma unroll
    for (int v = 0; v < 8; ++v) {
      int row = tm * 16 + v + ((lane < 16) ? 0 : 8);
      Y[(size_t)row * N + n] = f2bf(acc[v]);
    }
  }
}

// ---------------------------------------------------------------------------
// Kernel 2: fused attention per (b, h, 16-row q tile)
//   scores = Qh Kh^T / 32 -> LDS [16, PITCH] f32 -> rowwise exp + 1/sum
//   -> NT-stream normalized probs to sim_out (written exactly once)
//   -> P @ Vh with normalization folded into the output scale (P@V is linear)
// ---------------------------------------------------------------------------
__global__ __launch_bounds__(256)
void attn_fused_kernel(const __bf16* __restrict__ Qb,
                       const __bf16* __restrict__ Kb,
                       const __bf16* __restrict__ Vt,
                       float* __restrict__ attn_out,
                       float* __restrict__ sim_out) {
  __shared__ float sm[16 * PITCH];    // ~131 KB exp(score) tile, padded pitch
  __shared__ float red[4 * 8 * 32];   // 4 KB  P@V partial-sum exchange
  __shared__ float rowinv[16];        // per-row 1/sum(exp)

  const int lane = threadIdx.x & 31;
  const int wv   = threadIdx.x >> 5;

  const int bid = blockIdx.x;
  const int qt  = bid & 127;
  const int h   = (bid >> 7) & (N_HEADS - 1);
  const int b   = bid >> 11;

  const size_t headOff = ((size_t)(b * S_LEN + h * 128)) * D_MODEL;
  const __bf16* Qh = Qb + headOff;   // [2048, 64] row-major
  const __bf16* Kh = Kb + headOff;

  const int akb = (lane < 16) ? 0 : 16;
  const int bkb = (lane < 16) ? 0 : 16;

  // ---- Step 1: scaled scores -> LDS -----------------------------------------
  v16bf a0, a1;
  {
    const __bf16* qrow = Qh + (size_t)(qt * 16 + (lane & 15)) * D_HEAD;
    v8bf q0 = *(const v8bf*)(qrow + akb);
    v8bf q1 = *(const v8bf*)(qrow + 8 + akb);
    v8bf q2 = *(const v8bf*)(qrow + 32 + akb);
    v8bf q3 = *(const v8bf*)(qrow + 40 + akb);
#pragma unroll
    for (int i = 0; i < 8; ++i) {
      a0[i] = q0[i]; a0[8 + i] = q1[i];
      a1[i] = q2[i]; a1[8 + i] = q3[i];
    }
  }

  for (int kt = wv; kt < 128; kt += 8) {
    const __bf16* krow = Kh + (size_t)(kt * 16 + (lane & 15)) * D_HEAD;
    if (kt + 8 < 128)
      __builtin_prefetch(Kh + (size_t)((kt + 8) * 16 + (lane & 15)) * D_HEAD, 0, 0);
    v16bf b0 = *(const v16bf*)(krow + bkb);        // kk in [bkb, bkb+16)
    v16bf b1 = *(const v16bf*)(krow + 32 + bkb);   // kk in [32+bkb, 48+bkb)
    v8f acc = {};
    acc = __builtin_amdgcn_wmma_f32_16x16x32_bf16(false, a0, false, b0,
                                                  (short)0, acc, false, false);
    acc = __builtin_amdgcn_wmma_f32_16x16x32_bf16(false, a1, false, b1,
                                                  (short)0, acc, false, false);
#pragma unroll
    for (int v = 0; v < 8; ++v) {
      int r = v + ((lane < 16) ? 0 : 8);
      sm[r * PITCH + kt * 16 + (lane & 15)] = acc[v] * INV_SCALE;
    }
  }
  __syncthreads();

  // ---- Step 2: rowwise max, exp (stored unnormalized), 1/sum ----------------
  {
    int r  = threadIdx.x >> 4;
    int c0 = (threadIdx.x & 15) * 4;
    float* row = sm + r * PITCH;

    float mx = -3.4e38f;
    for (int c = c0; c < S_LEN; c += 64) {
      v4f x = *(const v4f*)(row + c);
      mx = fmaxf(mx, fmaxf(fmaxf(x[0], x[1]), fmaxf(x[2], x[3])));
    }
#pragma unroll
    for (int o = 8; o >= 1; o >>= 1) mx = fmaxf(mx, __shfl_xor(mx, o, 16));

    float sum = 0.f;
    for (int c = c0; c < S_LEN; c += 64) {
      v4f x = *(const v4f*)(row + c);
      v4f e;
      e[0] = __expf(x[0] - mx); e[1] = __expf(x[1] - mx);
      e[2] = __expf(x[2] - mx); e[3] = __expf(x[3] - mx);
      *(v4f*)(row + c) = e;
      sum += (e[0] + e[1]) + (e[2] + e[3]);
    }
#pragma unroll
    for (int o = 8; o >= 1; o >>= 1) sum += __shfl_xor(sum, o, 16);

    if ((threadIdx.x & 15) == 0) rowinv[r] = 1.0f / sum;
  }
  __syncthreads();

  // ---- Step 2b: NT-stream normalized probs to sim_out -----------------------
  {
    v4f* d4 = (v4f*)(sim_out +
                     ((size_t)((b * N_HEADS + h) * S_LEN + qt * 16)) * S_LEN);
    for (int i = threadIdx.x; i < 16 * (S_LEN / 4); i += 256) {
      int r = i >> 9;            // row (512 v4f per row)
      int c = i & 511;
      v4f x = *(const v4f*)(sm + r * PITCH + c * 4);
      __builtin_nontemporal_store(x * rowinv[r], d4 + (r << 9) + c);
    }
  }

  // ---- Step 3: P @ Vh -> 16x64 tile. 8 waves: 4 N-tiles x 2 K-halves --------
  {
    const int nt  = wv & 3;
    const int kh  = wv >> 2;                  // K half: [0,1024) or [1024,2048)
    const int col = nt * 16 + (lane & 15);    // column within head (0..63)
    const __bf16* vtrow =
        Vt + (((size_t)((b * N_HEADS + h) * 64 + col)) << 11);   // [64,2048] row
    const float* prow = sm + (lane & 15) * PITCH;

    v8f acc = {};
    for (int ks = kh * 32; ks < kh * 32 + 32; ++ks) {
      const float* ap = prow + ks * 32 + akb;
      v4f p0 = *(const v4f*)(ap);
      v4f p1 = *(const v4f*)(ap + 4);
      v4f p2 = *(const v4f*)(ap + 8);
      v4f p3 = *(const v4f*)(ap + 12);
      v16bf a;
#pragma unroll
      for (int i = 0; i < 4; ++i) {
        a[i]      = f2bf(p0[i]);
        a[4 + i]  = f2bf(p1[i]);
        a[8 + i]  = f2bf(p2[i]);
        a[12 + i] = f2bf(p3[i]);
      }
      v16bf bv = *(const v16bf*)(vtrow + ks * 32 + bkb);   // contiguous
      acc = __builtin_amdgcn_wmma_f32_16x16x32_bf16(false, a, false, bv,
                                                    (short)0, acc, false, false);
    }

    if (wv >= 4) {
#pragma unroll
      for (int v = 0; v < 8; ++v) red[(nt * 8 + v) * 32 + lane] = acc[v];
    }
    __syncthreads();
    if (wv < 4) {
#pragma unroll
      for (int v = 0; v < 8; ++v) {
        int r = v + ((lane < 16) ? 0 : 8);
        float out = (acc[v] + red[(nt * 8 + v) * 32 + lane]) * rowinv[r];
        int q = qt * 16 + r;                           // s2 index
        __builtin_nontemporal_store(
            out, &attn_out[((size_t)(b * S_LEN + q)) * D_MODEL + h * D_HEAD + col]);
      }
    }
  }
}

// ---------------------------------------------------------------------------
extern "C" void kernel_launch(void* const* d_in, const int* in_sizes, int n_in,
                              void* d_out, int out_size, void* d_ws, size_t ws_size,
                              hipStream_t stream) {
  const float* X  = (const float*)d_in[0];
  const float* Wq = (const float*)d_in[1];
  const float* Wk = (const float*)d_in[2];
  const float* Wv = (const float*)d_in[3];

  float* attn_out = (float*)d_out;                                  // [2,2048,1024]
  float* sim_out  = attn_out + (size_t)BATCH * S_LEN * D_MODEL;     // [2,16,2048,2048]

  const size_t elems = (size_t)BATCH * S_LEN * D_MODEL;   // 4,194,304
  __bf16* Qb = (__bf16*)d_ws;          // 8 MB each, 24 MB total in d_ws
  __bf16* Kb = Qb + elems;
  __bf16* Vt = Kb + elems;             // per-head transposed V [B*H, 64, 2048]

  // 3 * (4096/16) * (1024/16) = 49152 wave-tiles, 8 waves per 256-thread block
  qkv_proj_kernel<<<49152 / 8, 256, 0, stream>>>(X, Wq, Wk, Wv, Qb, Kb, Vt);

  // One block per (b, h, 16-row q tile): 2*16*128 = 4096 blocks
  attn_fused_kernel<<<BATCH * N_HEADS * (S_LEN / 16), 256, 0, stream>>>(
      Qb, Kb, Vt, attn_out, sim_out);
}
